// AttentionLSTM_8186207666734
// MI455X (gfx1250) — matrix operand order repaired and checked by reading, not compiled, not run on hardware
//
#include <hip/hip_runtime.h>
#include <hip/hip_bf16.h>
#include <math.h>

typedef __bf16 bf16;
typedef __attribute__((ext_vector_type(16))) __bf16 v16bf;
typedef __attribute__((ext_vector_type(8)))  __bf16 v8bf;
typedef __attribute__((ext_vector_type(8)))  float   v8f;

#define DEVI __device__ __forceinline__

// ---------------------------------------------------------------- utils
DEVI v8f zero8f() {
  v8f z;
#pragma unroll
  for (int i = 0; i < 8; ++i) z[i] = 0.0f;
  return z;
}
DEVI v16bf zero16bf() {
  v16bf z;
#pragma unroll
  for (int i = 0; i < 16; ++i) z[i] = (bf16)0.0f;
  return z;
}
DEVI float bflo(unsigned u) { return __uint_as_float(u << 16); }
DEVI float bfhi(unsigned u) { return __uint_as_float(u & 0xffff0000u); }
DEVI float sigmoidf(float x) { return 1.0f / (1.0f + expf(-x)); }

// ---------------------------------------------------------------- fragments
// A-matrix 16x32 bf16 fragment (row-major A, ld in elements).
// lanes 0-15 -> M=lane, elems 0..15 = K 0..15
//      lanes 16-31 -> M=lane-16, elems 0..15 = K 16..31
DEVI v16bf load_a_frag(const bf16* __restrict__ A, int ld, int row0, int kb, int lane) {
  const int half = lane >> 4;
  const bf16* q = A + (size_t)(row0 + (lane & 15)) * ld + kb + half * 16;
  v8bf lo = *(const v8bf*)q;
  v8bf hi = *(const v8bf*)(q + 8);
  v16bf r;
#pragma unroll
  for (int i = 0; i < 8; ++i) { r[i] = lo[i]; r[i + 8] = hi[i]; }
  return r;
}

// B fragment: Bt is N x K row-major (col n of B = row n of Bt, contiguous in K).
// Contiguous k (as A): lane holds col N=lane%16; lanes 0-15 K=kb..kb+15, lanes 16-31 K=kb+16..kb+31
DEVI v16bf load_b_frag(const bf16* __restrict__ Bt, int ld, int col0, int kb, int lane) {
  const bf16* q = Bt + (size_t)(col0 + (lane & 15)) * ld + kb + (lane >> 4) * 16;
  return *(const v16bf*)q;
}
DEVI v16bf load_b_frag_guard(const bf16* __restrict__ Bt, int ld, int col0, int kb,
                             int lane, int N) {
  const int r = col0 + (lane & 15);
  if (r >= N) return zero16bf();
  const bf16* q = Bt + (size_t)r * ld + kb + (lane >> 4) * 16;
  return *(const v16bf*)q;
}

// ---------------------------------------------------------------- big GEMM
// C[m][n] = sum_k A[m][k]*Bt[n][k] (+bias[n]). 8 waves, wave tile 64x32,
// block tile 128x128. Requires M%128==0, K%32==0.
template <bool OUT_BF16, bool NGUARD>
__global__ __launch_bounds__(256) void gemm_bf16_kernel(
    const bf16* __restrict__ A, int lda,
    const bf16* __restrict__ Bt, int ldb,
    void* __restrict__ Cout, int ldc,
    const float* __restrict__ bias,
    int N, int K) {
  const int lane = threadIdx.x & 31;
  const int wave = threadIdx.x >> 5;
  const int m0 = blockIdx.x * 128 + (wave & 1) * 64;
  const int n0 = blockIdx.y * 128 + (wave >> 1) * 32;

  v8f acc[4][2];
#pragma unroll
  for (int i = 0; i < 4; ++i)
#pragma unroll
    for (int j = 0; j < 2; ++j) acc[i][j] = zero8f();

  for (int kb = 0; kb < K; kb += 32) {
    v16bf a[4];
#pragma unroll
    for (int i = 0; i < 4; ++i) a[i] = load_a_frag(A, lda, m0 + 16 * i, kb, lane);
    v16bf b[2];
#pragma unroll
    for (int j = 0; j < 2; ++j)
      b[j] = NGUARD ? load_b_frag_guard(Bt, ldb, n0 + 16 * j, kb, lane, N)
                    : load_b_frag(Bt, ldb, n0 + 16 * j, kb, lane);
#pragma unroll
    for (int i = 0; i < 4; ++i)
#pragma unroll
      for (int j = 0; j < 2; ++j)
        acc[i][j] = __builtin_amdgcn_wmma_f32_16x16x32_bf16(
            false, a[i], false, b[j], (short)0, acc[i][j], false, false);
  }

  const int half = lane >> 4;
  const int ln = lane & 15;
#pragma unroll
  for (int j = 0; j < 2; ++j) {
    const int col = n0 + 16 * j + ln;
    if (NGUARD && col >= N) continue;
    const float bv = bias ? bias[col] : 0.0f;
#pragma unroll
    for (int i = 0; i < 4; ++i) {
      const int rowb = m0 + 16 * i + half * 8;
#pragma unroll
      for (int v = 0; v < 8; ++v) {
        const float val = acc[i][j][v] + bv;
        if (OUT_BF16)
          ((bf16*)Cout)[(size_t)(rowb + v) * ldc + col] = (bf16)val;
        else
          ((float*)Cout)[(size_t)(rowb + v) * ldc + col] = val;
      }
    }
  }
}

// ---------------------------------------------------------------- helpers
__global__ void cvt_f32_bf16_kernel(const float* __restrict__ src,
                                    bf16* __restrict__ dst, int n) {
  const int i = blockIdx.x * 256 + threadIdx.x;
  if (i < n) dst[i] = (bf16)src[i];
}

// Interleaved pack: row r = 4*j + g  <->  old row g*512 + j.
// Wp[r][0:512] = W_ih[g*512+j][0:512]; Wp[r][512:1024] = W_hh[g*512+j][:]
__global__ void pack_w_kernel(const float* __restrict__ Wih,
                              const float* __restrict__ Whh,
                              bf16* __restrict__ Wp) {
  const int i = blockIdx.x * 256 + threadIdx.x;  // over 2048*1024
  if (i >= 2048 * 1024) return;
  const int r = i >> 10, k = i & 1023;
  const int j = r >> 2, g = r & 3;
  const int srow = g * 512 + j;
  const float v = (k < 512) ? Wih[(size_t)srow * 7137 + k]
                            : Whh[(size_t)srow * 512 + (k - 512)];
  Wp[i] = (bf16)v;
}

__global__ void zero_u32_kernel(unsigned* __restrict__ p, int n) {
  const int i = blockIdx.x * 256 + threadIdx.x;
  if (i < n) p[i] = 0u;
}

// ---------------------------------------------------------------- attention (+ fused ph)
// One block per batch b:
//  ph[j]   = sum_k h[b,k]*h2h[j,k] + h2h_b[j]          (VALU, h2h L2-resident)
//  e[t]    = sum_h tanh(Hp[b,t,h]+ph[h])*sw[h]
//  alpha   = softmax_t(e)
//  ctx[d]  = sum_t alpha[t]*X[b,t,d]   -> ctx (bf16)
__global__ __launch_bounds__(256) void attn_ph_kernel(
    const bf16* __restrict__ Hp,      // [B*256, 512]
    const bf16* __restrict__ hprev,   // [B, 512]
    const bf16* __restrict__ h2h_bf,  // [512, 512]
    const float* __restrict__ h2h_b,  // [512]
    const float* __restrict__ sw,     // [512]
    const bf16* __restrict__ Xbf,     // [B*256, 512]
    bf16* __restrict__ ctx) {         // [B, 512]
  __shared__ float h_s[512], ph_s[512], e_s[256], red[256];
  const int b = blockIdx.x, tid = threadIdx.x;

  for (int j = tid; j < 512; j += 256) h_s[j] = (float)hprev[(size_t)b * 512 + j];
  __syncthreads();

  // ph row for this b (each thread: 2 output units, uint4 = 8 bf16 weights/iter)
  for (int j = tid; j < 512; j += 256) {
    float accp = h2h_b[j];
    const uint4* wrow = (const uint4*)(h2h_bf + (size_t)j * 512);
    for (int k4 = 0; k4 < 64; ++k4) {
      const uint4 u = wrow[k4];
      const int h0 = 8 * k4;
      accp += bflo(u.x) * h_s[h0 + 0] + bfhi(u.x) * h_s[h0 + 1];
      accp += bflo(u.y) * h_s[h0 + 2] + bfhi(u.y) * h_s[h0 + 3];
      accp += bflo(u.z) * h_s[h0 + 4] + bfhi(u.z) * h_s[h0 + 5];
      accp += bflo(u.w) * h_s[h0 + 6] + bfhi(u.w) * h_s[h0 + 7];
    }
    ph_s[j] = accp;
  }
  __syncthreads();

  // per-lane registers for its contiguous h-range [lane*16, lane*16+16)
  const int lane = tid & 31, wave = tid >> 5;
  float phr[16], swr[16];
#pragma unroll
  for (int i = 0; i < 16; ++i) {
    phr[i] = ph_s[lane * 16 + i];
    swr[i] = sw[lane * 16 + i];
  }

  // e[t]: two b128 loads per lane per t, 32-lane reduction
  for (int t = wave; t < 256; t += 8) {
    const v8bf* hp = (const v8bf*)(Hp + ((size_t)b * 256 + t) * 512 + lane * 16);
    const v8bf x0 = hp[0];
    const v8bf x1 = hp[1];
    float v = 0.0f;
#pragma unroll
    for (int i = 0; i < 8; ++i) v += tanhf((float)x0[i] + phr[i]) * swr[i];
#pragma unroll
    for (int i = 0; i < 8; ++i) v += tanhf((float)x1[i] + phr[8 + i]) * swr[8 + i];
#pragma unroll
    for (int off = 16; off > 0; off >>= 1) v += __shfl_xor(v, off, 32);
    if (lane == 0) e_s[t] = v;
  }
  __syncthreads();

  // softmax over T=256
  const float e = e_s[tid];
  red[tid] = e;
  __syncthreads();
  for (int off = 128; off > 0; off >>= 1) {
    if (tid < off) red[tid] = fmaxf(red[tid], red[tid + off]);
    __syncthreads();
  }
  const float mx = red[0];
  __syncthreads();
  const float ex = expf(e - mx);
  red[tid] = ex;
  __syncthreads();
  for (int off = 128; off > 0; off >>= 1) {
    if (tid < off) red[tid] += red[tid + off];
    __syncthreads();
  }
  const float inv = 1.0f / red[0];
  e_s[tid] = ex * inv;
  __syncthreads();

  // context: dword-pair loads (d = 2*tid, 2*tid+1)
  const unsigned* xp = (const unsigned*)(Xbf + (size_t)b * 256 * 512);
  float a0 = 0.0f, a1 = 0.0f;
  for (int t = 0; t < 256; ++t) {
    const unsigned u = xp[(size_t)t * 256 + tid];
    const float al = e_s[t];
    a0 += al * bflo(u);
    a1 += al * bfhi(u);
  }
  ctx[(size_t)b * 512 + 2 * tid] = (bf16)a0;
  ctx[(size_t)b * 512 + 2 * tid + 1] = (bf16)a1;
}

// ---------------------------------------------------------------- gates GEMM + LSTM (fused)
// gates[b][4j+g] = [ctx|h] @ Wp^T (rows interleaved 4j+g), staged in LDS,
// then epilogue applies biases + one-hot column gather + LSTM update.
// Grid: 32 blocks; block tile 128 rows x 64 cols (16 hidden units), 8 waves (2x4),
// wave tile 64x16.  h double-buffered (reads hprev, writes hnext).
__global__ __launch_bounds__(256) void gates_lstm_kernel(
    const bf16* __restrict__ ctx,     // [128, 512]   (K 0..511)
    const bf16* __restrict__ hprev,   // [128, 512]   (K 512..1023)
    const bf16* __restrict__ Wp,      // [2048, 1024] interleaved rows
    const float* __restrict__ b_ih,   // [2048]
    const float* __restrict__ b_hh,   // [2048]
    const float* __restrict__ W_ih,   // [2048, 7137]
    const int* __restrict__ targets,  // [128, 25]
    int s,
    float* __restrict__ c,            // [128, 512]
    bf16* __restrict__ hnext,         // [128, 512]
    bf16* __restrict__ hs) {          // [128*25, 512]
  __shared__ float clds[128][65];  // pad 1 -> conflict-free column reads
  const int lane = threadIdx.x & 31;
  const int wave = threadIdx.x >> 5;
  const int m0 = (wave & 1) * 64;
  const int nb = (wave >> 1) * 16;          // col offset inside block
  const int n0 = blockIdx.x * 64 + nb;      // global col (= 4j+g)

  v8f acc[4];
#pragma unroll
  for (int i = 0; i < 4; ++i) acc[i] = zero8f();

  for (int kb = 0; kb < 1024; kb += 32) {
    const bf16* Abase = (kb < 512) ? ctx : hprev;
    const int kk = (kb < 512) ? kb : kb - 512;
    v16bf a[4];
#pragma unroll
    for (int i = 0; i < 4; ++i) a[i] = load_a_frag(Abase, 512, m0 + 16 * i, kk, lane);
    const v16bf bfrag = load_b_frag(Wp, 1024, n0, kb, lane);
#pragma unroll
    for (int i = 0; i < 4; ++i)
      acc[i] = __builtin_amdgcn_wmma_f32_16x16x32_bf16(
          false, a[i], false, bfrag, (short)0, acc[i], false, false);
  }

  // stage tile to LDS
  const int half = lane >> 4;
  const int col = nb + (lane & 15);
#pragma unroll
  for (int i = 0; i < 4; ++i) {
    const int rowb = m0 + 16 * i + half * 8;
#pragma unroll
    for (int v = 0; v < 8; ++v) clds[rowb + v][col] = acc[i][v];
  }
  __syncthreads();

  // LSTM epilogue: 128 b x 16 hidden units per block
  for (int idx = threadIdx.x; idx < 2048; idx += 256) {
    const int b = idx & 127;
    const int jj = idx >> 7;                 // 0..15
    const int j = blockIdx.x * 16 + jj;      // global hidden unit
    const size_t gcol = 512 + (size_t)targets[b * 25 + s];
    const float iv = clds[b][4 * jj + 0] + b_ih[j] + b_hh[j] +
                     W_ih[(size_t)j * 7137 + gcol];
    const float fv = clds[b][4 * jj + 1] + b_ih[512 + j] + b_hh[512 + j] +
                     W_ih[(size_t)(512 + j) * 7137 + gcol];
    const float gv = clds[b][4 * jj + 2] + b_ih[1024 + j] + b_hh[1024 + j] +
                     W_ih[(size_t)(1024 + j) * 7137 + gcol];
    const float ov = clds[b][4 * jj + 3] + b_ih[1536 + j] + b_hh[1536 + j] +
                     W_ih[(size_t)(1536 + j) * 7137 + gcol];
    const float cn = sigmoidf(fv) * c[b * 512 + j] + sigmoidf(iv) * tanhf(gv);
    const float hn = sigmoidf(ov) * tanhf(cn);
    c[b * 512 + j] = cn;
    const bf16 hb = (bf16)hn;
    hnext[(size_t)b * 512 + j] = hb;
    hs[((size_t)b * 25 + s) * 512 + j] = hb;
  }
}

// ---------------------------------------------------------------- launch
extern "C" void kernel_launch(void* const* d_in, const int* in_sizes, int n_in,
                              void* d_out, int out_size, void* d_ws, size_t ws_size,
                              hipStream_t stream) {
  const float* inputs  = (const float*)d_in[0];   // [128,256,512]
  const int*   targets = (const int*)d_in[1];     // [128,25]
  const float* i2h_w   = (const float*)d_in[3];   // [512,512]
  const float* h2h_w   = (const float*)d_in[4];   // [512,512]
  const float* h2h_b   = (const float*)d_in[5];   // [512]
  const float* score_w = (const float*)d_in[6];   // [1,512]
  const float* W_ih    = (const float*)d_in[7];   // [2048,7137]
  const float* b_ih    = (const float*)d_in[8];   // [2048]
  const float* W_hh    = (const float*)d_in[9];   // [2048,512]
  const float* b_hh    = (const float*)d_in[10];  // [2048]
  const float* gen_w   = (const float*)d_in[11];  // [6625,512]
  const float* gen_b   = (const float*)d_in[12];  // [6625]
  float* out = (float*)d_out;                     // [128,25,6625]

  char* w = (char*)d_ws;
  auto alloc = [&](size_t bytes) {
    char* p = w;
    w += (bytes + 255) & ~(size_t)255;
    return p;
  };
  bf16* x_bf   = (bf16*)alloc(16777216ull * 2);  // inputs bf16
  bf16* hp_bf  = (bf16*)alloc(16777216ull * 2);  // H_proj bf16
  bf16* i2h_bf = (bf16*)alloc(262144ull * 2);
  bf16* h2h_bf = (bf16*)alloc(262144ull * 2);
  bf16* wp_bf  = (bf16*)alloc(2097152ull * 2);   // interleaved [W_ih_dense|W_hh]
  bf16* gw_bf  = (bf16*)alloc(3392000ull * 2);   // gen_w bf16
  bf16* ctx_bf = (bf16*)alloc(65536ull * 2);     // context per step
  bf16* h_a    = (bf16*)alloc(65536ull * 2);     // h ping
  bf16* h_b    = (bf16*)alloc(65536ull * 2);     // h pong
  bf16* hs_bf  = (bf16*)alloc(1638400ull * 2);   // all step outputs
  float* cbuf  = (float*)alloc(65536ull * 4);

  // --- one-time conversions / packing ---
  cvt_f32_bf16_kernel<<<(16777216 + 255) / 256, 256, 0, stream>>>(inputs, x_bf, 16777216);
  cvt_f32_bf16_kernel<<<1024, 256, 0, stream>>>(i2h_w, i2h_bf, 262144);
  cvt_f32_bf16_kernel<<<1024, 256, 0, stream>>>(h2h_w, h2h_bf, 262144);
  cvt_f32_bf16_kernel<<<(3392000 + 255) / 256, 256, 0, stream>>>(gen_w, gw_bf, 3392000);
  pack_w_kernel<<<(2097152 + 255) / 256, 256, 0, stream>>>(W_ih, W_hh, wp_bf);

  // --- state init (h0 = c0 = 0) ---
  zero_u32_kernel<<<(65536 + 255) / 256, 256, 0, stream>>>((unsigned*)cbuf, 65536);
  zero_u32_kernel<<<(32768 + 255) / 256, 256, 0, stream>>>((unsigned*)h_a, 32768);

  // --- H_proj = inputs @ i2h_w^T  (M=32768, N=512, K=512) -> bf16 ---
  gemm_bf16_kernel<true, false><<<dim3(256, 4), 256, 0, stream>>>(
      x_bf, 512, i2h_bf, 512, hp_bf, 512, nullptr, 512, 512);

  // --- recurrence over S=25 steps (h double-buffered) ---
  for (int s = 0; s < 25; ++s) {
    bf16* hprev = (s & 1) ? h_b : h_a;
    bf16* hnext = (s & 1) ? h_a : h_b;
    attn_ph_kernel<<<128, 256, 0, stream>>>(hp_bf, hprev, h2h_bf, h2h_b,
                                            score_w, x_bf, ctx_bf);
    gates_lstm_kernel<<<32, 256, 0, stream>>>(ctx_bf, hprev, wp_bf, b_ih, b_hh,
                                              W_ih, targets, s, cbuf, hnext, hs_bf);
  }

  // --- probs = hs @ gen_w^T + gen_b  (M=3200, N=6625, K=512) ---
  gemm_bf16_kernel<false, true><<<dim3(25, 52), 256, 0, stream>>>(
      hs_bf, 512, gw_bf, 512, out, 6625, gen_b, 6625, 512);
}
